// ClusterAtt_50740743635616
// MI455X (gfx1250) — matrix operand order repaired and checked by reading, not compiled, not run on hardware
//
#include <hip/hip_runtime.h>
#include <hip/hip_bf16.h>

typedef __bf16 v16bf __attribute__((ext_vector_type(16)));
typedef float  v8f  __attribute__((ext_vector_type(8)));
typedef int    v4i  __attribute__((vector_size(16)));

#if defined(__has_builtin)
#if __has_builtin(__builtin_amdgcn_global_load_async_to_lds_b128)
#define HAVE_ASYNC_LDS 1
#endif
#endif
#ifndef HAVE_ASYNC_LDS
#define HAVE_ASYNC_LDS 0
#endif

#define GLOBAL_AS __attribute__((address_space(1)))
#define LDS_AS    __attribute__((address_space(3)))

// ---------------- workspace layout (bytes) ----------------
#define WS_XB    ((size_t)0)            // ushort[8*256*16384]   67108864  (later reused as XASB)
#define WS_XTB   ((size_t)67108864)     // ushort[8*16384*256]   67108864
#define WS_XC    ((size_t)134217728)    // float [8*256*16384]  134217728  (later reused as XATT)
#define WS_GRAM  ((size_t)268435456)    // float [8*256*256]      2097152
#define WS_V     ((size_t)270532608)    // float [8*256*256]      2097152
#define WS_VB    ((size_t)272629760)    // ushort[8*256*256]      1048576
#define WS_XP    ((size_t)273678336)    // float [8*256*1024]     8388608
#define WS_XPTB  ((size_t)282066944)    // ushort[8*1024*256]     4194304
#define WS_QKV   ((size_t)286261248)    // float [8*768*1024]    25165824
#define WS_QSB   ((size_t)311427072)    // ushort[8*7*1024*32]    3670016
#define WS_KTB   ((size_t)315097088)    // ushort[8*7*1024*32]    3670016
#define WS_VTB   ((size_t)318767104)    // ushort[8*7*32*1024]    3670016
#define WS_XAP   ((size_t)322437120)    // float [8*224*1024]     7340032
#define WS_WQKV  ((size_t)329777152)    // ushort[768*256]         393216
#define WS_OUTW  ((size_t)330170368)    // ushort[256*224]         114688
#define WS_XATT  WS_XC                  // float [8*224*16384]  117440512 (reuse)
#define WS_XASB  WS_XB                  // ushort[8*16384*224]   58720256 (reuse)

#define OUT_XATT_N  33554432
#define OUT_NOISE   (33554432 + 1)

// ---------------- helpers ----------------
__device__ __forceinline__ unsigned short f2bf(float f) {
    unsigned int u = __float_as_uint(f);
    u += 0x7fffu + ((u >> 16) & 1u);
    return (unsigned short)(u >> 16);
}

__device__ __forceinline__ v8f vzero8() {
    v8f z = {0.f, 0.f, 0.f, 0.f, 0.f, 0.f, 0.f, 0.f};
    return z;
}

// async (or fallback sync) 16-byte global -> LDS transfer, per lane
__device__ __forceinline__ void stage16(const unsigned short* gsrc, unsigned short* ldst) {
#if HAVE_ASYNC_LDS
    __builtin_amdgcn_global_load_async_to_lds_b128(
        (GLOBAL_AS v4i*)gsrc,
        (LDS_AS v4i*)ldst, 0, 0);
#else
    *reinterpret_cast<uint4*>(ldst) = *reinterpret_cast<const uint4*>(gsrc);
#endif
}

__device__ __forceinline__ void stage_fence() {
#if HAVE_ASYNC_LDS
    asm volatile("s_wait_asynccnt 0" ::: "memory");
#endif
    __syncthreads();
}

// A-operand 16x32 bf16: lane = row (lane&15); contiguous 16 K at k0 + (lane hi ? 16 : 0)
__device__ __forceinline__ v16bf load_a(const unsigned short* base, int stride,
                                        int row0, int k0, int lane) {
    int r  = row0 + (lane & 15);
    int kb = k0 + ((lane & 16) ? 16 : 0);
    const unsigned short* p = base + (size_t)r * stride + kb;
    union { v16bf v; uint4 q[2]; } u;
    u.q[0] = *reinterpret_cast<const uint4*>(p);
    u.q[1] = *reinterpret_cast<const uint4*>(p + 8);
    return u.v;
}

// B-operand 32x16 bf16: lane = col (lane&15); contiguous 16 K at k0 + (lane hi ? 16 : 0)
// source array laid out [col][K] with K contiguous
__device__ __forceinline__ v16bf load_b(const unsigned short* base, int stride,
                                        int col0, int k0, int lane) {
    int c  = col0 + (lane & 15);
    int kb = k0 + ((lane & 16) ? 16 : 0);
    const unsigned short* p = base + (size_t)c * stride + kb;
    union { v16bf v; uint4 q[2]; } u;
    u.q[0] = *reinterpret_cast<const uint4*>(p);
    u.q[1] = *reinterpret_cast<const uint4*>(p + 8);
    return u.v;
}

__device__ __forceinline__ v8f wmma_bf16(v16bf a, v16bf b, v8f c) {
    return __builtin_amdgcn_wmma_f32_16x16x32_bf16(false, a, false, b, (short)0, c, false, false);
}

// ---------------- kernels ----------------

// x -> bf16 copy; also emit AVG_K scalar
__global__ void k_convert_x(const float* __restrict__ x, unsigned short* __restrict__ xb,
                            float* __restrict__ out) {
    int idx = blockIdx.x * 256 + threadIdx.x;
    xb[idx] = f2bf(x[idx]);
    if (idx == 0) out[OUT_XATT_N] = 8.0f;   // AVG_K
}

__global__ void k_convert_w(const float* __restrict__ qkv_w, const float* __restrict__ out_w,
                            unsigned short* __restrict__ wqkv, unsigned short* __restrict__ outwb) {
    int idx = blockIdx.x * 256 + threadIdx.x;
    if (idx < 196608)        wqkv[idx] = f2bf(qkv_w[idx]);
    else if (idx < 253952)   outwb[idx - 196608] = f2bf(out_w[idx - 196608]);
}

// x[b][c][n] -> xtb[b][n][c] (bf16), 32x32 LDS tiles
__global__ void k_transpose(const float* __restrict__ x, unsigned short* __restrict__ xtb) {
    __shared__ float tile[32][33];
    int b = blockIdx.z, ct = blockIdx.y, nt = blockIdx.x;
    int c0 = ct * 32, n0 = nt * 32;
    int tx = threadIdx.x & 31, ty = threadIdx.x >> 5;
    const float* X = x + (size_t)b * 256 * 16384;
    for (int yy = ty; yy < 32; yy += 8)
        tile[yy][tx] = X[(size_t)(c0 + yy) * 16384 + n0 + tx];
    __syncthreads();
    unsigned short* XT = xtb + (size_t)b * 16384 * 256;
    for (int yy = ty; yy < 32; yy += 8)
        XT[(size_t)(n0 + yy) * 256 + c0 + tx] = f2bf(tile[tx][yy]);
}

// gram = X X^T per batch. Workgroup (4 waves) computes a 64x64 block; the 64 A-rows
// and 64 B-rows per K-step are staged once in LDS (async global->LDS when available)
// and shared by all 4 waves.
__global__ void k_gram(const unsigned short* __restrict__ xb, float* __restrict__ gram) {
    __shared__ unsigned short As[64 * 32];
    __shared__ unsigned short Bs[64 * 32];
    int wid = threadIdx.x >> 5, lane = threadIdx.x & 31;
    int blk = blockIdx.x;            // 8 batches * 16 blocks
    int b = blk >> 4, t = blk & 15;
    int rb0 = (t >> 2) * 64, cb0 = (t & 3) * 64;
    const unsigned short* X = xb + (size_t)b * 256 * 16384;
    v8f acc0 = vzero8(), acc1 = vzero8(), acc2 = vzero8(), acc3 = vzero8();
    int rA = (lane & 15);
    int kbA = (lane & 16) ? 16 : 0;
    int kbB = (lane & 16) ? 16 : 0;
    for (int k = 0; k < 16384; k += 32) {
        // stage 64x32 A-rows and 64x32 B-rows (8KB) cooperatively
        for (int j = 0; j < 2; j++) {
            int u = threadIdx.x + j * 128;
            int row = u >> 2, q4 = (u & 3) * 8;
            stage16(X + (size_t)(rb0 + row) * 16384 + k + q4, &As[row * 32 + q4]);
            stage16(X + (size_t)(cb0 + row) * 16384 + k + q4, &Bs[row * 32 + q4]);
        }
        stage_fence();
        union { v16bf v; uint4 q[2]; } ua;
        ua.q[0] = *reinterpret_cast<const uint4*>(&As[(wid * 16 + rA) * 32 + kbA]);
        ua.q[1] = *reinterpret_cast<const uint4*>(&As[(wid * 16 + rA) * 32 + 8 + kbA]);
        v16bf A = ua.v;
        v8f* accs[4] = {&acc0, &acc1, &acc2, &acc3};
        for (int j = 0; j < 4; j++) {
            union { v16bf v; uint4 q[2]; } ub;
            int cB = j * 16 + (lane & 15);
            ub.q[0] = *reinterpret_cast<const uint4*>(&Bs[cB * 32 + kbB]);
            ub.q[1] = *reinterpret_cast<const uint4*>(&Bs[cB * 32 + kbB + 8]);
            *accs[j] = wmma_bf16(A, ub.v, *accs[j]);
        }
        __syncthreads();
    }
    float* G = gram + (size_t)b * 65536;
    int col = lane & 15, rbl = (lane & 16) ? 8 : 0;
    v8f* accs[4] = {&acc0, &acc1, &acc2, &acc3};
    for (int j = 0; j < 4; j++)
        for (int i = 0; i < 8; i++) {
            int r = rb0 + wid * 16 + rbl + i;
            G[r * 256 + cb0 + j * 16 + col] = (*accs[j])[i];
        }
}

// parallel-ordered cyclic Jacobi eigensolver; one workgroup per batch.
// gram buffer is destroyed (becomes ~diagonal). V rows accumulate Q^T.
// Output: Vb = rows of Q^T sorted by ascending eigenvalue, bf16.
__global__ void k_jacobi(float* __restrict__ gram, float* __restrict__ V,
                         unsigned short* __restrict__ Vb) {
    int b = blockIdx.x;
    float* A  = gram + (size_t)b * 65536;
    float* Vv = V    + (size_t)b * 65536;
    int t = threadIdx.x;
    __shared__ float cs_c[128], cs_s[128];
    __shared__ int   cs_p[128], cs_q[128];
    __shared__ float ev[256];
    __shared__ int   pidx[256];

    for (int i = t; i < 65536; i += 256)
        Vv[i] = ((i >> 8) == (i & 255)) ? 1.f : 0.f;
    __syncthreads();

    const int nm1 = 255;
    for (int sweep = 0; sweep < 6; sweep++) {
        for (int r = 0; r < nm1; r++) {
            if (t < 128) {
                int sA = t, sB = 255 - t;
                int p = (sA == 0) ? 0 : 1 + ((sA - 1 + r) % nm1);
                int q = 1 + ((sB - 1 + r) % nm1);
                if (p > q) { int tmp = p; p = q; q = tmp; }
                float app = A[p * 256 + p], aqq = A[q * 256 + q], apq = A[p * 256 + q];
                float c = 1.f, s = 0.f;
                if (fabsf(apq) > 1e-12f * (fabsf(app) + fabsf(aqq)) + 1e-30f) {
                    float tau = (aqq - app) / (2.f * apq);
                    float tt  = (tau >= 0.f ? 1.f : -1.f) / (fabsf(tau) + sqrtf(1.f + tau * tau));
                    c = 1.f / sqrtf(1.f + tt * tt);
                    s = tt * c;
                }
                cs_c[t] = c; cs_s[t] = s; cs_p[t] = p; cs_q[t] = q;
            }
            __syncthreads();
            // A <- A J  (disjoint column pairs, no write conflicts)
            for (int idx = t; idx < 128 * 256; idx += 256) {
                int k = idx >> 8, i = idx & 255;
                int p = cs_p[k], q = cs_q[k];
                float c = cs_c[k], s = cs_s[k];
                float aip = A[i * 256 + p], aiq = A[i * 256 + q];
                A[i * 256 + p] = c * aip - s * aiq;
                A[i * 256 + q] = s * aip + c * aiq;
            }
            __syncthreads();
            // A <- J^T A ; V <- J^T V (disjoint row pairs)
            for (int idx = t; idx < 128 * 256; idx += 256) {
                int k = idx >> 8, j = idx & 255;
                int p = cs_p[k], q = cs_q[k];
                float c = cs_c[k], s = cs_s[k];
                float apj = A[p * 256 + j], aqj = A[q * 256 + j];
                A[p * 256 + j] = c * apj - s * aqj;
                A[q * 256 + j] = s * apj + c * aqj;
                float vpj = Vv[p * 256 + j], vqj = Vv[q * 256 + j];
                Vv[p * 256 + j] = c * vpj - s * vqj;
                Vv[q * 256 + j] = s * vpj + c * vqj;
            }
            __syncthreads();
        }
    }
    // sort eigenvalues ascending (odd-even transposition on (val, idx))
    ev[t] = A[t * 256 + t];
    pidx[t] = t;
    __syncthreads();
    for (int pass = 0; pass < 256; pass++) {
        int off = pass & 1;
        if (t < 128) {
            int i = 2 * t + off;
            if (i + 1 < 256 && ev[i] > ev[i + 1]) {
                float tv = ev[i]; ev[i] = ev[i + 1]; ev[i + 1] = tv;
                int ti = pidx[i]; pidx[i] = pidx[i + 1]; pidx[i + 1] = ti;
            }
        }
        __syncthreads();
    }
    unsigned short* VB = Vb + (size_t)b * 65536;
    for (int i = t; i < 65536; i += 256) {
        int rr = i >> 8, cc = i & 255;
        VB[i] = f2bf(Vv[pidx[rr] * 256 + cc]);
    }
}

// x_cluster = V_sorted @ X + X ; also write noise slice (c<32) to d_out.
// Workgroup computes 64 channels x 64 pixels; xtb slab staged once in LDS,
// shared by all 4 channel-strip waves (cuts xtb L2 reads 4x).
__global__ void k_xproj(const unsigned short* __restrict__ vb, const unsigned short* __restrict__ xtb,
                        const float* __restrict__ x, float* __restrict__ xc,
                        float* __restrict__ noise_out) {
    __shared__ unsigned short Bs[64 * 32];
    int wid = threadIdx.x >> 5, lane = threadIdx.x & 31;
    int blk = blockIdx.x;               // 8 * 1024
    int b = blk >> 10, t = blk & 1023;
    int c0 = (t >> 8) * 64;             // channel block
    int n0 = (t & 255) * 64;            // pixel block
    int r0 = c0 + wid * 16;             // wave's channel strip
    const unsigned short* Vb_ = vb  + (size_t)b * 65536;
    const unsigned short* XT  = xtb + (size_t)b * 16384 * 256;
    v8f acc0 = vzero8(), acc1 = vzero8(), acc2 = vzero8(), acc3 = vzero8();
    int kbB = (lane & 16) ? 16 : 0;
    for (int k = 0; k < 256; k += 32) {
        for (int j = 0; j < 2; j++) {
            int u = threadIdx.x + j * 128;
            int row = u >> 2, q4 = (u & 3) * 8;
            stage16(XT + (size_t)(n0 + row) * 256 + k + q4, &Bs[row * 32 + q4]);
        }
        stage_fence();
        v16bf A = load_a(Vb_, 256, r0, k, lane);
        v8f* accs[4] = {&acc0, &acc1, &acc2, &acc3};
        for (int j = 0; j < 4; j++) {
            union { v16bf v; uint4 q[2]; } ub;
            int cB = j * 16 + (lane & 15);
            ub.q[0] = *reinterpret_cast<const uint4*>(&Bs[cB * 32 + kbB]);
            ub.q[1] = *reinterpret_cast<const uint4*>(&Bs[cB * 32 + kbB + 8]);
            *accs[j] = wmma_bf16(A, ub.v, *accs[j]);
        }
        __syncthreads();
    }
    const float* Xb_ = x  + (size_t)b * 256 * 16384;
    float*       XC  = xc + (size_t)b * 256 * 16384;
    int col = lane & 15, rbl = (lane & 16) ? 8 : 0;
    v8f* accs[4] = {&acc0, &acc1, &acc2, &acc3};
    for (int j = 0; j < 4; j++)
        for (int i = 0; i < 8; i++) {
            int c  = r0 + rbl + i;
            int nn = n0 + j * 16 + col;
            float val = (*accs[j])[i] + Xb_[(size_t)c * 16384 + nn];
            XC[(size_t)c * 16384 + nn] = val;
            if (c < 32) noise_out[((size_t)b * 32 + c) * 16384 + nn] = val;
        }
}

// 4x4 maxpool of x_cluster -> xp (f32, [b][c][1024]) and xptb (bf16, [b][1024][c])
__global__ void k_maxpool(const float* __restrict__ xc, float* __restrict__ xp,
                          unsigned short* __restrict__ xptb) {
    int idx = blockIdx.x * 256 + threadIdx.x;   // 8*256*1024
    int b = idx >> 18, rem = idx & 262143;
    int c = rem >> 10, p = rem & 1023;
    int ph = p >> 5, pw = p & 31;
    const float* S = xc + ((size_t)b * 256 + c) * 16384;
    float m = -1e30f;
    for (int dy = 0; dy < 4; dy++)
        for (int dx = 0; dx < 4; dx++)
            m = fmaxf(m, S[(ph * 4 + dy) * 128 + pw * 4 + dx]);
    xp[((size_t)b * 256 + c) * 1024 + p] = m;
    xptb[((size_t)b * 1024 + p) * 256 + c] = f2bf(m);
}

// qkv = W(768x256) @ Xp(256x1024) + bias, per batch
__global__ void k_qkv(const unsigned short* __restrict__ wqkv, const unsigned short* __restrict__ xptb,
                      const float* __restrict__ qkv_bias, float* __restrict__ qkv) {
    int gw   = blockIdx.x * 4 + (threadIdx.x >> 5);
    int lane = threadIdx.x & 31;
    int b = gw / 768, t = gw % 768;
    int r0 = (t >> 4) * 16;        // output-channel tile (0..767)
    int p0 = (t & 15) * 64;        // 64-wide pixel group
    const unsigned short* XPT = xptb + (size_t)b * 1024 * 256;
    v8f acc0 = vzero8(), acc1 = vzero8(), acc2 = vzero8(), acc3 = vzero8();
    for (int k = 0; k < 256; k += 32) {
        v16bf A = load_a(wqkv, 256, r0, k, lane);
        acc0 = wmma_bf16(A, load_b(XPT, 256, p0,      k, lane), acc0);
        acc1 = wmma_bf16(A, load_b(XPT, 256, p0 + 16, k, lane), acc1);
        acc2 = wmma_bf16(A, load_b(XPT, 256, p0 + 32, k, lane), acc2);
        acc3 = wmma_bf16(A, load_b(XPT, 256, p0 + 48, k, lane), acc3);
    }
    float* Q = qkv + (size_t)b * 768 * 1024;
    int col = lane & 15, rb = (lane & 16) ? 8 : 0;
    v8f* accs[4] = {&acc0, &acc1, &acc2, &acc3};
    for (int j = 0; j < 4; j++)
        for (int i = 0; i < 8; i++) {
            int o = r0 + rb + i;
            Q[(size_t)o * 1024 + p0 + j * 16 + col] = (*accs[j])[i] + qkv_bias[o];
        }
}

// build attention operands: qsb[b][n][q][c] (q+emb scaled), ktb[b][n][p][c]=K^T, vtb[b][n][c][p]=V^T
__global__ void k_attnprep(const float* __restrict__ qkv, const float* __restrict__ emb,
                           unsigned short* __restrict__ qsb, unsigned short* __restrict__ ktb,
                           unsigned short* __restrict__ vtb) {
    int idx = blockIdx.x * 256 + threadIdx.x;   // 8*7*1024*32
    int b = idx / 229376, rem = idx % 229376;
    int n = rem / 32768, rem2 = rem % 32768;
    int r = rem2 >> 5, c2 = rem2 & 31;
    const float* QP = qkv + (size_t)b * 768 * 1024;
    size_t base = (((size_t)b * 7 + n) * 1024 + r) * 32 + c2;
    float qv = QP[(size_t)(32 + n * 32 + (r >> 5)) * 1024 + (r & 31) * 32 + c2]
             + emb[(((size_t)n * 8 + b) * 1024 + r) * 32 + c2];
    qsb[base] = f2bf(qv * 0.17677669529663687f);   // fold 1/sqrt(32)
    float kv = QP[(size_t)256 * 1024 + (size_t)(32 + n * 32 + c2) * 1024 + r];
    ktb[base] = f2bf(kv);
    float vv = QP[(size_t)512 * 1024 + (size_t)(32 + n * 32 + (r >> 5)) * 1024 + (r & 31) * 32 + c2];
    vtb[(((size_t)b * 7 + n) * 32 + c2) * 1024 + r] = f2bf(vv);
}

// flash-style attention: one wave = 16 q rows; online softmax over p in chunks of 32
__global__ void k_attention(const unsigned short* __restrict__ qsb, const unsigned short* __restrict__ ktb,
                            const unsigned short* __restrict__ vtb, const float* __restrict__ xp,
                            float* __restrict__ xap) {
    int wid = threadIdx.x >> 5, lane = threadIdx.x & 31;
    int blk = blockIdx.x;                 // 8*7*16
    int b = blk / 112, rem = blk % 112;
    int n = rem / 16, qg = rem % 16;
    int q0 = qg * 64 + wid * 16;
    const unsigned short* Q  = qsb + (((size_t)b * 7 + n) * 1024) * 32;
    const unsigned short* KT = ktb + (((size_t)b * 7 + n) * 1024) * 32;
    const unsigned short* VT = vtb + (((size_t)b * 7 + n) * 32) * 1024;
    __shared__ unsigned short plds[4 * 512];
    int wbase = wid * 512;

    v16bf a = load_a(Q, 32, q0, 0, lane);
    v8f o0 = vzero8(), o1 = vzero8();
    float m[8], l[8];
    for (int i = 0; i < 8; i++) { m[i] = -1e30f; l[i] = 0.f; }
    int col = lane & 15, rb = (lane & 16) ? 8 : 0;

    for (int p0 = 0; p0 < 1024; p0 += 32) {
        v16bf kb0 = load_b(KT, 32, p0,      0, lane);
        v16bf kb1 = load_b(KT, 32, p0 + 16, 0, lane);
        v8f s0 = wmma_bf16(a, kb0, vzero8());
        v8f s1 = wmma_bf16(a, kb1, vzero8());
        float pe0[8], pe1[8];
        for (int i = 0; i < 8; i++) {
            float mx = fmaxf(s0[i], s1[i]);
            for (int off = 8; off >= 1; off >>= 1) mx = fmaxf(mx, __shfl_xor(mx, off, 32));
            float mn = fmaxf(m[i], mx);
            float sc = __expf(m[i] - mn);
            float e0 = __expf(s0[i] - mn);
            float e1 = __expf(s1[i] - mn);
            float rs = e0 + e1;
            for (int off = 8; off >= 1; off >>= 1) rs += __shfl_xor(rs, off, 32);
            l[i] = l[i] * sc + rs;
            m[i] = mn;
            o0[i] *= sc; o1[i] *= sc;
            pe0[i] = e0; pe1[i] = e1;
        }
        // stage P (16x32) to LDS in C-layout, re-read in A-layout
        for (int i = 0; i < 8; i++) {
            int row = rb + i;
            plds[wbase + row * 32 + col]      = f2bf(pe0[i]);
            plds[wbase + row * 32 + 16 + col] = f2bf(pe1[i]);
        }
        asm volatile("s_wait_dscnt 0" ::: "memory");
        int r  = lane & 15;
        int kb = (lane & 16) ? 16 : 0;
        union { v16bf v; uint4 q[2]; } u;
        u.q[0] = *reinterpret_cast<const uint4*>(&plds[wbase + r * 32 + kb]);
        u.q[1] = *reinterpret_cast<const uint4*>(&plds[wbase + r * 32 + 8 + kb]);
        v16bf ap  = u.v;
        v16bf vb0 = load_b(VT, 1024, 0,  p0, lane);
        v16bf vb1 = load_b(VT, 1024, 16, p0, lane);
        o0 = wmma_bf16(ap, vb0, o0);
        o1 = wmma_bf16(ap, vb1, o1);
    }

    const float* XPb = xp  + (size_t)b * 256 * 1024;
    float*       XAP = xap + (size_t)b * 224 * 1024;
    for (int i = 0; i < 8; i++) {
        float inv = 1.f / l[i];
        int qrow = q0 + rb + i;
        int ch = n * 32 + (qrow >> 5);
        int sp = (qrow & 31) * 32;
        XAP[(size_t)ch * 1024 + sp + col]      = o0[i] * inv + XPb[(size_t)(32 + ch) * 1024 + sp + col];
        XAP[(size_t)ch * 1024 + sp + 16 + col] = o1[i] * inv + XPb[(size_t)(32 + ch) * 1024 + sp + 16 + col];
    }
}

// bilinear 4x upsample (half-pixel, edge clamp): xap (b,224,32,32) -> xatt (b,224,128,128)
__global__ void k_upsample(const float* __restrict__ xap, float* __restrict__ xatt) {
    int idx = blockIdx.x * 256 + threadIdx.x;   // 8*224*16384
    int b = idx / 3670016, rem = idx % 3670016;
    int c = rem >> 14, pix = rem & 16383;
    int y = pix >> 7, xq = pix & 127;
    float sy = y * 0.25f - 0.375f;
    float sx = xq * 0.25f - 0.375f;
    float fy = sy - floorf(sy), fx = sx - floorf(sx);
    int y0 = (int)floorf(sy), x0 = (int)floorf(sx);
    int y0c = min(max(y0, 0), 31),     y1c = min(max(y0 + 1, 0), 31);
    int x0c = min(max(x0, 0), 31),     x1c = min(max(x0 + 1, 0), 31);
    const float* S = xap + ((size_t)b * 224 + c) * 1024;
    float v = (1.f - fy) * ((1.f - fx) * S[y0c * 32 + x0c] + fx * S[y0c * 32 + x1c])
            +        fy  * ((1.f - fx) * S[y1c * 32 + x0c] + fx * S[y1c * 32 + x1c]);
    xatt[((size_t)b * 224 + c) * 16384 + pix] = v;
}

// noise_proj on the fly, cosine similarity over channels, scale, store bf16 pixel-major
__global__ void k_simscale(const float* __restrict__ xatt, const float* __restrict__ noise,
                           const float* __restrict__ sim_w, unsigned short* __restrict__ xasb) {
    int idx = blockIdx.x * 256 + threadIdx.x;   // 8*16384
    int b = idx >> 14, pix = idx & 16383;
    const float* NZ = noise + (size_t)b * 32 * 16384 + pix;
    float nz[32];
    for (int k = 0; k < 32; k++) nz[k] = NZ[(size_t)k * 16384];
    const float* XA = xatt + (size_t)b * 224 * 16384 + pix;
    float sxx = 0.f, snn = 0.f, sxn = 0.f;
    for (int c = 0; c < 224; c++) {
        const float* w = sim_w + c * 32;
        float np = 0.f;
        for (int k = 0; k < 32; k++) np += w[k] * nz[k];
        float xa = XA[(size_t)c * 16384];
        sxx += xa * xa; snn += np * np; sxn += xa * np;
    }
    float nx = fmaxf(sqrtf(sxx), 1e-12f);
    float nn = fmaxf(sqrtf(snn), 1e-12f);
    float sim = (sxn / (nx * nn) + 1.f) * 0.5f;
    float scale = 1.f - sim;
    unsigned short* XO = xasb + ((size_t)b * 16384 + pix) * 224;
    for (int c = 0; c < 224; c++) XO[c] = f2bf(scale * XA[(size_t)c * 16384]);
}

// final conv1x1: y[b][o][pix] = sum_c out_w[o][c] * xas[b][pix][c] (computed as y^T tiles)
__global__ void k_outgemm(const unsigned short* __restrict__ xasb, const unsigned short* __restrict__ outwb,
                          float* __restrict__ out) {
    int gw   = blockIdx.x * 4 + (threadIdx.x >> 5);
    int lane = threadIdx.x & 31;
    int b = gw >> 12, t = gw & 4095;
    int pix0 = (t >> 2) * 16;
    int o0   = (t & 3) * 64;
    const unsigned short* XA = xasb + (size_t)b * 16384 * 224;
    v8f acc0 = vzero8(), acc1 = vzero8(), acc2 = vzero8(), acc3 = vzero8();
    for (int k = 0; k < 224; k += 32) {
        v16bf A = load_a(XA, 224, pix0, k, lane);
        acc0 = wmma_bf16(A, load_b(outwb, 224, o0,      k, lane), acc0);
        acc1 = wmma_bf16(A, load_b(outwb, 224, o0 + 16, k, lane), acc1);
        acc2 = wmma_bf16(A, load_b(outwb, 224, o0 + 32, k, lane), acc2);
        acc3 = wmma_bf16(A, load_b(outwb, 224, o0 + 48, k, lane), acc3);
    }
    int col = lane & 15, rb = (lane & 16) ? 8 : 0;
    v8f* accs[4] = {&acc0, &acc1, &acc2, &acc3};
    for (int j = 0; j < 4; j++)
        for (int i = 0; i < 8; i++) {
            int pixr = pix0 + rb + i;
            int o = o0 + j * 16 + col;
            out[((size_t)b * 256 + o) * 16384 + pixr] = (*accs[j])[i];
        }
}

// ---------------- launch ----------------
extern "C" void kernel_launch(void* const* d_in, const int* in_sizes, int n_in,
                              void* d_out, int out_size, void* d_ws, size_t ws_size,
                              hipStream_t stream) {
    const float* x      = (const float*)d_in[0];
    const float* qkv_w  = (const float*)d_in[1];
    const float* qkv_b  = (const float*)d_in[2];
    const float* emb    = (const float*)d_in[3];
    const float* sim_w  = (const float*)d_in[4];
    const float* out_w  = (const float*)d_in[5];
    float* out = (float*)d_out;
    char*  ws  = (char*)d_ws;

    unsigned short* xb    = (unsigned short*)(ws + WS_XB);
    unsigned short* xtb   = (unsigned short*)(ws + WS_XTB);
    float*          xc    = (float*)(ws + WS_XC);
    float*          gram  = (float*)(ws + WS_GRAM);
    float*          V     = (float*)(ws + WS_V);
    unsigned short* vb    = (unsigned short*)(ws + WS_VB);
    float*          xp    = (float*)(ws + WS_XP);
    unsigned short* xptb  = (unsigned short*)(ws + WS_XPTB);
    float*          qkv   = (float*)(ws + WS_QKV);
    unsigned short* qsb   = (unsigned short*)(ws + WS_QSB);
    unsigned short* ktb   = (unsigned short*)(ws + WS_KTB);
    unsigned short* vtb   = (unsigned short*)(ws + WS_VTB);
    float*          xap   = (float*)(ws + WS_XAP);
    unsigned short* wqkv  = (unsigned short*)(ws + WS_WQKV);
    unsigned short* outwb = (unsigned short*)(ws + WS_OUTW);
    float*          xatt  = (float*)(ws + WS_XATT);   // reuses xc region
    unsigned short* xasb  = (unsigned short*)(ws + WS_XASB); // reuses xb region
    float*          noise_out = out + OUT_NOISE;

    k_convert_x<<<131072, 256, 0, stream>>>(x, xb, out);
    k_convert_w<<<992, 256, 0, stream>>>(qkv_w, out_w, wqkv, outwb);
    k_transpose<<<dim3(512, 8, 8), 256, 0, stream>>>(x, xtb);
    k_gram<<<128, 128, 0, stream>>>(xb, gram);
    k_jacobi<<<8, 256, 0, stream>>>(gram, V, vb);
    k_xproj<<<8192, 128, 0, stream>>>(vb, xtb, x, xc, noise_out);
    k_maxpool<<<8192, 256, 0, stream>>>(xc, xp, xptb);
    k_qkv<<<1536, 128, 0, stream>>>(wqkv, xptb, qkv_b, qkv);
    k_attnprep<<<7168, 256, 0, stream>>>(qkv, emb, qsb, ktb, vtb);
    k_attention<<<896, 128, 0, stream>>>(qsb, ktb, vtb, xp, xap);
    k_upsample<<<114688, 256, 0, stream>>>(xap, xatt);
    k_simscale<<<512, 256, 0, stream>>>(xatt, noise_out, sim_w, xasb);
    k_outgemm<<<8192, 128, 0, stream>>>(xasb, outwb, out);
    (void)in_sizes; (void)n_in; (void)out_size; (void)ws_size;
}
